// DeepIRT_47785806135744
// MI455X (gfx1250) — matrix operand order repaired and checked by reading, compiled, not run on hardware
//
#include <hip/hip_runtime.h>
#include <hip/hip_bf16.h>
#include <math.h>
#include <stdint.h>

#define NSKILL 1000
#define D_ 128
#define M_ 64
#define B_ 64
#define L_ 512
#define ROWS (B_ * L_)   // 32768
#define SEG_ 16
#define TSEG_ (L_ / SEG_)  // 32

typedef __bf16 bf16;
typedef __attribute__((ext_vector_type(16))) __bf16 v16bf;
typedef __attribute__((ext_vector_type(8)))  __bf16 bf16x8;
typedef __attribute__((ext_vector_type(8)))  float  v8f;
typedef __attribute__((ext_vector_type(4)))  int    v4i;

// ------------------------------------------------ optional CDNA5 async LDS path
#if defined(__has_builtin)
#if __has_builtin(__builtin_amdgcn_global_load_async_to_lds_b128) && \
    __has_builtin(__builtin_amdgcn_s_wait_asynccnt)
#define USE_ASYNC_LDS 1
#endif
#endif

#ifdef USE_ASYNC_LDS
typedef __attribute__((address_space(1))) v4i* as1_v4i_p;
typedef __attribute__((address_space(3))) v4i* as3_v4i_p;
#endif

// stage one w-row (64 floats, 256 B) into LDS
__device__ __forceinline__ void stage_w(const float* __restrict__ g, float* dst, int tid) {
#ifdef USE_ASYNC_LDS
  if (tid < 16) {
    __builtin_amdgcn_global_load_async_to_lds_b128(
        (as1_v4i_p)(uintptr_t)(g + tid * 4),
        (as3_v4i_p)(dst + tid * 4), 0, 0);
  }
#else
  if (tid < M_) dst[tid] = g[tid];
#endif
}
__device__ __forceinline__ void stage_sync() {
#ifdef USE_ASYNC_LDS
  __builtin_amdgcn_s_wait_asynccnt(0);
#endif
  __syncthreads();
}

// ---------------------------------------------------------------- WMMA helpers

__device__ __forceinline__ v8f wmma_bf16(v16bf a, v16bf b, v8f c) {
  return __builtin_amdgcn_wmma_f32_16x16x32_bf16(
      false, a, false, b, (short)0, c, false, false);
}

// A-fragment: lane 0-15 row=r0+lane, K = kk0+{0..7,16..23}; lanes 16-31 K = kk0+{8..15,24..31}
__device__ __forceinline__ v16bf frag_a_ptr(const bf16* p) {
  bf16x8 c0 = *(const bf16x8*)(p);
  bf16x8 c1 = *(const bf16x8*)(p + 16);
  v16bf f;
#pragma unroll
  for (int i = 0; i < 8; ++i) { f[i] = c0[i]; f[i + 8] = c1[i]; }
  return f;
}
__device__ __forceinline__ v16bf frag_a(const bf16* base, int ld, int r0, int kk0, int lane) {
  int r  = r0 + (lane & 15);
  int kb = kk0 + ((lane & 16) ? 8 : 0);
  return frag_a_ptr(base + (size_t)r * ld + kb);
}

// B-fragment from N-major weights WT[n][kk]: lane 0-15 K = kk0+{0..7,16..23}; lanes 16-31 K = kk0+{8..15,24..31}
__device__ __forceinline__ v16bf frag_b(const bf16* baseT, int ld, int n0, int kk0, int lane) {
  int n  = n0 + (lane & 15);
  int kb = kk0 + ((lane & 16) ? 8 : 0);
  const bf16* p = baseT + (size_t)n * ld + kb;
  bf16x8 c0 = *(const bf16x8*)(p);
  bf16x8 c1 = *(const bf16x8*)(p + 16);
  v16bf f;
#pragma unroll
  for (int i = 0; i < 8; ++i) { f[i] = c0[i]; f[i + 8] = c1[i]; }
  return f;
}

__device__ __forceinline__ float sigmoidf_(float x) { return 1.0f / (1.0f + expf(-x)); }

// ---------------------------------------------------------------- 1) gather

__global__ void k_gather(const int* __restrict__ skills, const int* __restrict__ resp,
                         const float* __restrict__ k_emb, const float* __restrict__ v_emb,
                         bf16* __restrict__ kbf, bf16* __restrict__ vbf,
                         float* __restrict__ kg) {
  int i = blockIdx.x * blockDim.x + threadIdx.x;
  if (i >= ROWS * D_) return;
  int r = i >> 7, c = i & 127;
  int s  = skills[r];
  int rr = resp[r];
  int mr = (rr > -1) ? rr : 0;
  int x  = s + NSKILL * mr;
  float kv = k_emb[(size_t)s * D_ + c];
  float vv = v_emb[(size_t)x * D_ + c];
  kbf[i] = (bf16)kv;
  kg[i]  = kv;
  vbf[i] = (bf16)vv;
}

// ---------------------------------------------------------------- 2) weight convert/transpose

__global__ void k_weights(const float* __restrict__ Mk, const float* __restrict__ eW,
                          const float* __restrict__ aW, const float* __restrict__ fW,
                          bf16* __restrict__ Mkbf, bf16* __restrict__ eWT,
                          bf16* __restrict__ aWT, bf16* __restrict__ fWT) {
  int i = blockIdx.x * blockDim.x + threadIdx.x;
  if (i < M_ * D_) Mkbf[i] = (bf16)Mk[i];
  if (i < D_ * D_) {
    int kk = i >> 7, n = i & 127;
    eWT[n * D_ + kk] = (bf16)eW[i];
    aWT[n * D_ + kk] = (bf16)aW[i];
  }
  if (i < 2 * D_ * D_) {
    int kk = i >> 7, n = i & 127;
    fWT[n * (2 * D_) + kk] = (bf16)fW[i];
  }
}

// ---------------------------------------------------------------- 3) w = softmax(k @ Mk^T)

__global__ void k_wsoftmax(const bf16* __restrict__ kbf, const bf16* __restrict__ Mkbf,
                           float* __restrict__ w) {
  int lane = threadIdx.x;
  int r0 = blockIdx.x * 16;
  v16bf A[4];
#pragma unroll
  for (int ks = 0; ks < 4; ++ks) A[ks] = frag_a(kbf, D_, r0, ks * 32, lane);
  v8f C[4];
#pragma unroll
  for (int nt = 0; nt < 4; ++nt) {
    v8f c = {};
#pragma unroll
    for (int ks = 0; ks < 4; ++ks)
      c = wmma_bf16(A[ks], frag_b(Mkbf, D_, nt * 16, ks * 32, lane), c);
    C[nt] = c;
  }
#pragma unroll
  for (int j = 0; j < 8; ++j) {
    float m = C[0][j];
#pragma unroll
    for (int nt = 1; nt < 4; ++nt) m = fmaxf(m, C[nt][j]);
#pragma unroll
    for (int msk = 1; msk < 16; msk <<= 1) m = fmaxf(m, __shfl_xor(m, msk, 32));
    float ex[4], s = 0.f;
#pragma unroll
    for (int nt = 0; nt < 4; ++nt) { ex[nt] = expf(C[nt][j] - m); s += ex[nt]; }
#pragma unroll
    for (int msk = 1; msk < 16; msk <<= 1) s += __shfl_xor(s, msk, 32);
    float inv = 1.0f / s;
    int grow = r0 + j + ((lane & 16) ? 8 : 0);
#pragma unroll
    for (int nt = 0; nt < 4; ++nt)
      w[(size_t)grow * M_ + nt * 16 + (lane & 15)] = ex[nt] * inv;
  }
}

// ---------------------------------------------------------------- 4) e, a

__global__ void k_ea(const bf16* __restrict__ vbf, const bf16* __restrict__ eWT,
                     const bf16* __restrict__ aWT, const float* __restrict__ eb,
                     const float* __restrict__ ab, float* __restrict__ e,
                     float* __restrict__ a) {
  int lane = threadIdx.x;
  int r0 = blockIdx.x * 16;
  v16bf A[4];
#pragma unroll
  for (int ks = 0; ks < 4; ++ks) A[ks] = frag_a(vbf, D_, r0, ks * 32, lane);
#pragma unroll
  for (int nt = 0; nt < 8; ++nt) {
    int n = nt * 16 + (lane & 15);
    v8f ce = {}, ca = {};
#pragma unroll
    for (int ks = 0; ks < 4; ++ks) {
      ce = wmma_bf16(A[ks], frag_b(eWT, D_, nt * 16, ks * 32, lane), ce);
      ca = wmma_bf16(A[ks], frag_b(aWT, D_, nt * 16, ks * 32, lane), ca);
    }
    float ebn = eb[n], abn = ab[n];
#pragma unroll
    for (int j = 0; j < 8; ++j) {
      int grow = r0 + j + ((lane & 16) ? 8 : 0);
      e[(size_t)grow * D_ + n] = sigmoidf_(ce[j] + ebn);
      a[(size_t)grow * D_ + n] = tanhf(ca[j] + abn);
    }
  }
}

// ---------------------------------------------------------------- 5a) segmented scan pass 1
// per (b, seg): fold TSEG_ steps into per-element linear transfer Mv_out = A*Mv_in + Bc
// element (m,d):  alpha = 1 - w[m]*e[d], beta = w[m]*a[d]

__device__ __forceinline__ size_t seg_idx(int s, int b, int m, int d) {
  return (((size_t)s * B_ + b) * M_ + m) * D_ + d;
}

__global__ void __launch_bounds__(256) k_scan_seg(const float* __restrict__ w,
                                                  const float* __restrict__ e,
                                                  const float* __restrict__ a,
                                                  float* __restrict__ Aseg,
                                                  float* __restrict__ Bseg) {
  __shared__ float wS[2][M_];
  int tid = threadIdx.x;
  int b = blockIdx.x, s = blockIdx.y;
  int d = tid & 127, mh = tid >> 7;
  float Ar[32], Br[32];
#pragma unroll
  for (int j = 0; j < 32; ++j) { Ar[j] = 1.f; Br[j] = 0.f; }
  size_t row0 = (size_t)b * L_ + s * TSEG_;
  stage_w(w + row0 * M_, wS[0], tid);
  stage_sync();
  for (int t = 0; t < TSEG_; ++t) {
    size_t row = row0 + t;
    float ed = e[row * D_ + d];
    float ad = a[row * D_ + d];
    if (t + 1 < TSEG_) stage_w(w + (row + 1) * M_, wS[(t + 1) & 1], tid);
    const float* wb = wS[t & 1] + mh * 32;
#pragma unroll
    for (int j = 0; j < 32; ++j) {
      float wj = wb[j];
      float al = fmaf(-wj, ed, 1.0f);   // 1 - w*e
      float be = wj * ad;               // w*a
      Ar[j] = Ar[j] * al;
      Br[j] = fmaf(Br[j], al, be);
    }
    stage_sync();
  }
#pragma unroll
  for (int j = 0; j < 32; ++j) {
    size_t idx = seg_idx(s, b, mh * 32 + j, d);
    Aseg[idx] = Ar[j];
    Bseg[idx] = Br[j];
  }
}

// ---------------------------------------------------------------- 5b) serial segment combine

__global__ void __launch_bounds__(256) k_scan_mid(const float* __restrict__ Aseg,
                                                  const float* __restrict__ Bseg,
                                                  const float* __restrict__ Mv0,
                                                  float* __restrict__ St) {
  int tid = threadIdx.x;
  int b = blockIdx.x;
  int d = tid & 127, mh = tid >> 7;
  float mv[32];
#pragma unroll
  for (int j = 0; j < 32; ++j) mv[j] = Mv0[(size_t)(mh * 32 + j) * D_ + d];
  for (int s = 0; s < SEG_; ++s) {
#pragma unroll
    for (int j = 0; j < 32; ++j) {
      size_t idx = seg_idx(s, b, mh * 32 + j, d);
      St[idx] = mv[j];
      mv[j] = fmaf(mv[j], Aseg[idx], Bseg[idx]);
    }
  }
}

// ---------------------------------------------------------------- 5c) segmented scan pass 2

__global__ void __launch_bounds__(256) k_scan_emit(const float* __restrict__ w,
                                                   const float* __restrict__ e,
                                                   const float* __restrict__ a,
                                                   const float* __restrict__ St,
                                                   bf16* __restrict__ readsbf) {
  __shared__ float wS[2][M_];
  __shared__ float partS[2][D_];
  int tid = threadIdx.x;
  int b = blockIdx.x, s = blockIdx.y;
  int d = tid & 127, mh = tid >> 7;
  float mv[32];
#pragma unroll
  for (int j = 0; j < 32; ++j) mv[j] = St[seg_idx(s, b, mh * 32 + j, d)];
  size_t row0 = (size_t)b * L_ + s * TSEG_;
  stage_w(w + row0 * M_, wS[0], tid);
  stage_sync();
  for (int t = 0; t < TSEG_; ++t) {
    size_t row = row0 + t;
    float ed = e[row * D_ + d];
    float ad = a[row * D_ + d];
    if (t + 1 < TSEG_) stage_w(w + (row + 1) * M_, wS[(t + 1) & 1], tid);
    const float* wb = wS[t & 1] + mh * 32;
    float part = 0.f;
#pragma unroll
    for (int j = 0; j < 32; ++j) {
      float wj = wb[j];
      part = fmaf(wj, mv[j], part);        // read uses PRE-update memory
      float t1 = fmaf(-mv[j], ed, ad);
      mv[j] = fmaf(wj, t1, mv[j]);
    }
    if (mh) partS[t & 1][d] = part;
    stage_sync();
    if (!mh) readsbf[row * D_ + d] = (bf16)(part + partS[t & 1][d]);
  }
}

// ---------------------------------------------------------------- 5-fallback) sequential scan

__global__ void __launch_bounds__(256) k_scan(const float* __restrict__ w,
                                              const float* __restrict__ e,
                                              const float* __restrict__ a,
                                              const float* __restrict__ Mv0,
                                              bf16* __restrict__ readsbf) {
  __shared__ float wS[M_];
  __shared__ float partS[D_];
  int tid = threadIdx.x;
  int b   = blockIdx.x;
  int d   = tid & 127;
  int mh  = tid >> 7;
  float mv[32];
#pragma unroll
  for (int j = 0; j < 32; ++j) mv[j] = Mv0[(size_t)(mh * 32 + j) * D_ + d];
  for (int t = 0; t < L_; ++t) {
    size_t row = (size_t)b * L_ + t;
    if (tid < M_) wS[tid] = w[row * M_ + tid];
    float ed = e[row * D_ + d];
    float ad = a[row * D_ + d];
    __syncthreads();
    float part = 0.f;
#pragma unroll
    for (int j = 0; j < 32; ++j) {
      float wj = wS[mh * 32 + j];
      part = fmaf(wj, mv[j], part);
      float t1 = fmaf(-mv[j], ed, ad);
      mv[j] = fmaf(wj, t1, mv[j]);
    }
    if (mh) partS[d] = part;
    __syncthreads();
    if (!mh) readsbf[row * D_ + d] = (bf16)(part + partS[d]);
  }
}

// ---------------------------------------------------------------- 6) que_diff

__global__ void k_qdiff(const float* __restrict__ kg, const float* __restrict__ dW,
                        const float* __restrict__ db, float* __restrict__ qd) {
  int gt   = blockIdx.x * blockDim.x + threadIdx.x;
  int wave = gt >> 5;
  int lane = threadIdx.x & 31;
  if (wave >= ROWS) return;
  const float4 kv = ((const float4*)(kg + (size_t)wave * D_))[lane];
  const float4 dv = ((const float4*)dW)[lane];
  float s = kv.x * dv.x + kv.y * dv.y + kv.z * dv.z + kv.w * dv.w;
#pragma unroll
  for (int msk = 16; msk >= 1; msk >>= 1) s += __shfl_xor(s, msk, 32);
  if (lane == 0) qd[wave] = tanhf(s + db[0]);
}

// ---------------------------------------------------------------- 7) f GEMM + heads

__global__ void k_fhead(const bf16* __restrict__ readsbf, const bf16* __restrict__ kbf,
                        const bf16* __restrict__ fWT, const float* __restrict__ fb,
                        const float* __restrict__ abW, const float* __restrict__ abb,
                        const float* __restrict__ qd, float* __restrict__ out) {
  int lane = threadIdx.x;
  int r0 = blockIdx.x * 16;
  v16bf A[8];
#pragma unroll
  for (int ks = 0; ks < 8; ++ks) {
    int kk0 = ks * 32;
    int kb  = kk0 + ((lane & 16) ? 8 : 0);
    int r   = r0 + (lane & 15);
    const bf16* p = (kk0 < D_) ? (readsbf + (size_t)r * D_ + kb)
                               : (kbf + (size_t)r * D_ + (kb - D_));
    A[ks] = frag_a_ptr(p);
  }
  float acc[8];
#pragma unroll
  for (int j = 0; j < 8; ++j) acc[j] = 0.f;
#pragma unroll
  for (int nt = 0; nt < 8; ++nt) {
    int n = nt * 16 + (lane & 15);
    v8f c = {};
#pragma unroll
    for (int ks = 0; ks < 8; ++ks)
      c = wmma_bf16(A[ks], frag_b(fWT, 2 * D_, nt * 16, ks * 32, lane), c);
    float fbn = fb[n], awn = abW[n];
#pragma unroll
    for (int j = 0; j < 8; ++j)
      acc[j] = fmaf(tanhf(c[j] + fbn), awn, acc[j]);
  }
#pragma unroll
  for (int j = 0; j < 8; ++j) {
    float s = acc[j];
#pragma unroll
    for (int msk = 1; msk < 16; msk <<= 1) s += __shfl_xor(s, msk, 32);
    if ((lane & 15) == 0) {
      int grow = r0 + j + ((lane & 16) ? 8 : 0);
      float stu = tanhf(s + abb[0]);
      float p   = sigmoidf_(3.0f * stu - qd[grow]);
      int bb = grow >> 9, l = grow & 511;
      if (l >= 1) out[(size_t)bb * (L_ - 1) + (l - 1)] = p;
    }
  }
}

// ---------------------------------------------------------------- launch

extern "C" void kernel_launch(void* const* d_in, const int* in_sizes, int n_in,
                              void* d_out, int out_size, void* d_ws, size_t ws_size,
                              hipStream_t stream) {
  const int*   skills = (const int*)d_in[0];
  const int*   resp   = (const int*)d_in[1];
  const float* k_emb  = (const float*)d_in[2];
  const float* v_emb  = (const float*)d_in[3];
  const float* Mk     = (const float*)d_in[4];
  const float* Mv0    = (const float*)d_in[5];
  const float* fW     = (const float*)d_in[6];
  const float* fb     = (const float*)d_in[7];
  const float* eW     = (const float*)d_in[8];
  const float* eb     = (const float*)d_in[9];
  const float* aW     = (const float*)d_in[10];
  const float* ab     = (const float*)d_in[11];
  const float* abW    = (const float*)d_in[12];
  const float* abb    = (const float*)d_in[13];
  const float* dW     = (const float*)d_in[14];
  const float* db     = (const float*)d_in[15];
  float* out = (float*)d_out;

  char* ws = (char*)d_ws;
  size_t off = 0;
  auto carve = [&](size_t bytes) -> void* {
    void* p = ws + off;
    off = (off + bytes + 255) & ~(size_t)255;
    return p;
  };
  bf16*  kbf     = (bf16*)carve((size_t)ROWS * D_ * sizeof(bf16));
  bf16*  vbf     = (bf16*)carve((size_t)ROWS * D_ * sizeof(bf16));
  float* kg      = (float*)carve((size_t)ROWS * D_ * sizeof(float));
  float* w       = (float*)carve((size_t)ROWS * M_ * sizeof(float));
  float* e       = (float*)carve((size_t)ROWS * D_ * sizeof(float));
  float* a       = (float*)carve((size_t)ROWS * D_ * sizeof(float));
  bf16*  readsbf = (bf16*)carve((size_t)ROWS * D_ * sizeof(bf16));
  float* qd      = (float*)carve((size_t)ROWS * sizeof(float));
  bf16*  Mkbf    = (bf16*)carve((size_t)M_ * D_ * sizeof(bf16));
  bf16*  eWT     = (bf16*)carve((size_t)D_ * D_ * sizeof(bf16));
  bf16*  aWT     = (bf16*)carve((size_t)D_ * D_ * sizeof(bf16));
  bf16*  fWT     = (bf16*)carve((size_t)2 * D_ * D_ * sizeof(bf16));
  // segmented-scan buffers (optional, large): A, B transfers + segment start states
  size_t seg_elems = (size_t)SEG_ * B_ * M_ * D_;
  float* Aseg = (float*)carve(seg_elems * sizeof(float));
  float* Bseg = (float*)carve(seg_elems * sizeof(float));
  float* St   = (float*)carve(seg_elems * sizeof(float));
  bool use_seg = (off <= ws_size);
  (void)in_sizes; (void)n_in; (void)out_size;

  int tiles = ROWS / 16;  // 2048

  k_gather<<<(ROWS * D_ + 255) / 256, 256, 0, stream>>>(skills, resp, k_emb, v_emb,
                                                        kbf, vbf, kg);
  k_weights<<<(2 * D_ * D_ + 255) / 256, 256, 0, stream>>>(Mk, eW, aW, fW,
                                                           Mkbf, eWT, aWT, fWT);
  k_wsoftmax<<<tiles, 32, 0, stream>>>(kbf, Mkbf, w);
  k_ea<<<tiles, 32, 0, stream>>>(vbf, eWT, aWT, eb, ab, e, a);
  k_qdiff<<<(ROWS * 32 + 255) / 256, 256, 0, stream>>>(kg, dW, db, qd);
  if (use_seg) {
    dim3 g(B_, SEG_);
    k_scan_seg<<<g, 256, 0, stream>>>(w, e, a, Aseg, Bseg);
    k_scan_mid<<<B_, 256, 0, stream>>>(Aseg, Bseg, Mv0, St);
    k_scan_emit<<<g, 256, 0, stream>>>(w, e, a, St, readsbf);
  } else {
    k_scan<<<B_, 256, 0, stream>>>(w, e, a, Mv0, readsbf);
  }
  k_fhead<<<tiles, 32, 0, stream>>>(readsbf, kbf, fWT, fb, abW, abb, qd, out);
}
